// Mamba2_80814104642067
// MI455X (gfx1250) — matrix operand order repaired and checked by reading, not compiled, not run on hardware
//
#include <hip/hip_runtime.h>

typedef __attribute__((ext_vector_type(16))) __bf16 v16bf;
typedef __attribute__((ext_vector_type(8)))  __bf16 v8bf;
typedef __attribute__((ext_vector_type(8)))  float  v8f;

#define DEVINL __device__ __forceinline__

// ---- problem constants (from reference) ----
constexpr int NL = 2, DM = 1024, DI = 2048, H = 32, P = 64, Ns = 64, G = 1, DC = 4, CS = 256;
constexpr int Bb = 2, L = 2048;
constexpr int DIN   = 2 * DI + 2 * G * Ns + H;   // 4256
constexpr int CONVD = DI + 2 * G * Ns;           // 2176
constexpr int NCH   = L / CS;                    // 8
constexpr int ROWS  = Bb * L;                    // 4096
constexpr float EPS = 1e-5f;

// ---------------- WMMA helpers ----------------
DEVINL v16bf load_a16(const __bf16* rowp, int kb) {
  v8bf lo = *(const v8bf*)(rowp + 2 * kb);
  v8bf hi = *(const v8bf*)(rowp + 2 * kb + 8);
  v16bf a;
#pragma unroll
  for (int j = 0; j < 8; ++j) { a[j] = lo[j]; a[j + 8] = hi[j]; }
  return a;
}

DEVINL v8f wmma_bf16(v16bf a, v16bf b, v8f c) {
  return __builtin_amdgcn_wmma_f32_16x16x32_bf16(false, a, false, b, (short)0, c, false, false);
}

DEVINL float siluf(float x) { return x / (1.0f + expf(-x)); }

// LDS byte-offset of a generic pointer known to point into shared memory
typedef __attribute__((address_space(3))) __bf16 lds_bf16_t;
DEVINL unsigned lds_off(__bf16* p) {
  return (unsigned)(unsigned long long)(lds_bf16_t*)p;
}

// async 16B copy global -> LDS (per-lane), tracked by ASYNCcnt
DEVINL void async_ld_b128(__bf16* lds_dst, const __bf16* gsrc) {
  unsigned l = lds_off(lds_dst);
  unsigned long long g = (unsigned long long)gsrc;
  asm volatile("global_load_async_to_lds_b128 %0, %1, off" :: "v"(l), "v"(g) : "memory");
}
DEVINL void wait_async_le8() { asm volatile("s_wait_asynccnt 0x8" ::: "memory"); }
DEVINL void wait_async_0()   { asm volatile("s_wait_asynccnt 0x0" ::: "memory"); }

// ---------------- elementwise convert ----------------
__global__ void f32_to_bf16_k(const float* __restrict__ in, __bf16* __restrict__ out, int n) {
  int i = blockIdx.x * 256 + threadIdx.x;
  if (i < n) out[i] = (__bf16)in[i];
}

// ---------------- GEMM: C[M x N] = X[M x K](bf16) * W[N x K]^T(bf16), f32 out ----------------
// 256 thr (8 waves), block tile 128x128, K-slab 64, double-buffered async LDS staging.
// wave tile 32x64 (2x4 WMMA accums).
constexpr int LDA_S = 72;   // LDS row pitch (elements), 64 + 8 pad -> bank spread

template<int ACC>
__global__ void __launch_bounds__(256)
gemm_xwt(const __bf16* __restrict__ X, const __bf16* __restrict__ W,
         float* __restrict__ C, int Ncols, int K) {
  __shared__ __bf16 As[2][128 * LDA_S];
  __shared__ __bf16 Bs[2][128 * LDA_S];

  const int tid  = threadIdx.x;
  const int lane = tid & 31;
  const int w    = tid >> 5;
  const int wm   = w >> 1;                 // 0..3
  const int wn   = w & 1;                  // 0..1
  const int mBlk = blockIdx.y * 128;
  const int nBlk = blockIdx.x * 128;
  const int lh = lane & 15;
  const int kb = (lane < 16) ? 0 : 8;
  const int kh = (lane < 16) ? 0 : 16;
  const int NK = K >> 6;                   // number of 64-wide K slabs

  // stage one 128x64 bf16 tile per matrix: 1024 chunks of 16B, 4 per thread
  auto issue = [&](int buf, int k0) {
#pragma unroll
    for (int q = 0; q < 4; ++q) {
      const int e = tid + q * 256;         // 0..1023
      const int row = e >> 3, c8 = e & 7;
      async_ld_b128(&As[buf][row * LDA_S + c8 * 8],
                    X + (size_t)(mBlk + row) * K + k0 + c8 * 8);
    }
#pragma unroll
    for (int q = 0; q < 4; ++q) {
      const int e = tid + q * 256;
      const int row = e >> 3, c8 = e & 7;
      async_ld_b128(&Bs[buf][row * LDA_S + c8 * 8],
                    W + (size_t)(nBlk + row) * K + k0 + c8 * 8);
    }
  };

  v8f acc[2][4] = {};

  issue(0, 0);
  for (int s = 0; s < NK; ++s) {
    const int cur = s & 1;
    if (s + 1 < NK) { issue(1 - cur, (s + 1) * 64); wait_async_le8(); }
    else           { wait_async_0(); }
    __syncthreads();

    const __bf16* Ab = &As[cur][0];
    const __bf16* Bv = &Bs[cur][0];
#pragma unroll
    for (int ks = 0; ks < 64; ks += 32) {
      v16bf afrag[2];
#pragma unroll
      for (int mi = 0; mi < 2; ++mi)
        afrag[mi] = load_a16(&Ab[(wm * 32 + mi * 16 + lh) * LDA_S + ks], kb);
#pragma unroll
      for (int ni = 0; ni < 4; ++ni) {
        v16bf bfrag = *(const v16bf*)&Bv[(wn * 64 + ni * 16 + lh) * LDA_S + ks + kh];
#pragma unroll
        for (int mi = 0; mi < 2; ++mi)
          acc[mi][ni] = wmma_bf16(afrag[mi], bfrag, acc[mi][ni]);
      }
    }
    __syncthreads();   // everyone done reading buf[cur] before it is refilled
  }

  const int rbase = (lane < 16) ? 0 : 8;
#pragma unroll
  for (int mi = 0; mi < 2; ++mi)
#pragma unroll
    for (int ni = 0; ni < 4; ++ni) {
      const int col = nBlk + wn * 64 + ni * 16 + lh;
#pragma unroll
      for (int r = 0; r < 8; ++r) {
        const int row = mBlk + wm * 32 + mi * 16 + r + rbase;
        float* cp = C + (size_t)row * Ncols + col;
        if (ACC) *cp += acc[mi][ni][r];
        else     *cp  = acc[mi][ni][r];
      }
    }
}

// ---------------- RMSNorm(DM) -> bf16 ----------------
__global__ void __launch_bounds__(256)
rmsnorm_to_bf16(const float* __restrict__ x, const float* __restrict__ wgt,
                __bf16* __restrict__ out) {
  const int row = blockIdx.x;
  const float* xr = x + (size_t)row * DM;
  __shared__ float red[256];
  float s = 0.f;
  for (int i = threadIdx.x; i < DM; i += 256) { float v = xr[i]; s += v * v; }
  red[threadIdx.x] = s; __syncthreads();
  for (int off = 128; off > 0; off >>= 1) {
    if (threadIdx.x < off) red[threadIdx.x] += red[threadIdx.x + off];
    __syncthreads();
  }
  const float scale = rsqrtf(red[0] / (float)DM + EPS);
  for (int i = threadIdx.x; i < DM; i += 256)
    out[(size_t)row * DM + i] = (__bf16)(xr[i] * scale * wgt[i]);
}

// ---------------- depthwise conv (DC=4, causal) + SiLU ----------------
__global__ void conv_silu_k(const float* __restrict__ zx, const float* __restrict__ cw,
                            const float* __restrict__ cb, float* __restrict__ xBC) {
  int idx = blockIdx.x * 256 + threadIdx.x;
  if (idx >= ROWS * CONVD) return;
  const int ch = idx % CONVD;
  const int row = idx / CONVD;
  const int b = row / L, l = row % L;
  float s = cb[ch];
#pragma unroll
  for (int t = 0; t < DC; ++t) {
    const int lp = l - (DC - 1) + t;
    if (lp >= 0) s += zx[(size_t)(b * L + lp) * DIN + DI + ch] * cw[ch * DC + t];
  }
  xBC[(size_t)row * CONVD + ch] = siluf(s);
}

// ---------------- dt = softplus(raw + bias); cum = cumsum(dt * -exp(A_log)) per chunk ----------------
__global__ void __launch_bounds__(256)
dt_cum_k(const float* __restrict__ zx, const float* __restrict__ dt_bias,
         const float* __restrict__ A_log, float* __restrict__ dt_g, float* __restrict__ cum) {
  const int blk = blockIdx.x;
  const int h = blk % H;
  const int c = (blk / H) % NCH;
  const int b = blk / (H * NCH);
  const int i = threadIdx.x;
  const int row = b * L + c * CS + i;
  const float v = zx[(size_t)row * DIN + DI + CONVD + h] + dt_bias[h];
  const float dtv = (v > 20.f) ? v : log1pf(expf(v));
  dt_g[(size_t)row * H + h] = dtv;
  const float dA = -expf(A_log[h]) * dtv;
  __shared__ float s[256];
  s[i] = dA; __syncthreads();
  for (int off = 1; off < 256; off <<= 1) {
    const float t = (i >= off) ? s[i - off] : 0.f;
    __syncthreads();
    s[i] += t;
    __syncthreads();
  }
  cum[((size_t)(b * H + h) * NCH + c) * CS + i] = s[i];
}

// ---------------- per-chunk states: states[p,n] = sum_j Xdt[p,j] * (B[j,n]*decay[j]) ----------------
__global__ void __launch_bounds__(128)
states_k(const float* __restrict__ xBC, const float* __restrict__ dt_g,
         const float* __restrict__ cum, float* __restrict__ states) {
  const int blk = blockIdx.x;
  const int h = blk % H;
  const int c = (blk / H) % NCH;
  const int b = blk / (H * NCH);
  __shared__ __bf16 XdtT[P * CS];   // [p][j]
  __shared__ __bf16 BdT[Ns * CS];   // [n][j]
  const float* cumb = cum + ((size_t)(b * H + h) * NCH + c) * CS;
  const float cl = cumb[CS - 1];
  for (int j = threadIdx.x; j < CS; j += 128) {
    const int row = b * L + c * CS + j;
    const float dtv = dt_g[(size_t)row * H + h];
    const float dec = expf(cl - cumb[j]);
    const float* xr = xBC + (size_t)row * CONVD;
    for (int p = 0; p < P; ++p)  XdtT[p * CS + j] = (__bf16)(xr[h * P + p] * dtv);
    for (int n = 0; n < Ns; ++n) BdT[n * CS + j]  = (__bf16)(xr[DI + n] * dec);
  }
  __syncthreads();
  const int tid = threadIdx.x, lane = tid & 31, w = tid >> 5;
  const int lh = lane & 15, kb = (lane < 16) ? 0 : 8, kh = (lane < 16) ? 0 : 16;
  const int rbase = (lane < 16) ? 0 : 8;
  const int pm = w * 16;
  v8f acc[4] = {};
  for (int k0 = 0; k0 < CS; k0 += 32) {
    v16bf a = load_a16(&XdtT[(pm + lh) * CS + k0], kb);
#pragma unroll
    for (int ni = 0; ni < 4; ++ni) {
      v16bf bf = *(const v16bf*)&BdT[(ni * 16 + lh) * CS + k0 + kh];
      acc[ni] = wmma_bf16(a, bf, acc[ni]);
    }
  }
  float* outp = states + ((size_t)(b * NCH + c) * H + h) * (P * Ns);
#pragma unroll
  for (int ni = 0; ni < 4; ++ni)
#pragma unroll
    for (int r = 0; r < 8; ++r)
      outp[(pm + r + rbase) * Ns + ni * 16 + lh] = acc[ni][r];
}

// ---------------- sequential inter-chunk scan (8 steps) ----------------
__global__ void __launch_bounds__(256)
scan_k(const float* __restrict__ states, const float* __restrict__ cum,
       float* __restrict__ prev) {
  const int h = blockIdx.x % H;
  const int b = blockIdx.x / H;
  const int e0 = threadIdx.x * 16;
  float acc[16];
#pragma unroll
  for (int k = 0; k < 16; ++k) acc[k] = 0.f;
  for (int c = 0; c < NCH; ++c) {
    const float dec = expf(cum[((size_t)(b * H + h) * NCH + c) * CS + CS - 1]);
    const size_t base = ((size_t)(b * NCH + c) * H + h) * (P * Ns);
#pragma unroll
    for (int k = 0; k < 16; ++k) {
      prev[base + e0 + k] = acc[k];
      acc[k] = acc[k] * dec + states[base + e0 + k];
    }
  }
}

// ---------------- intra-chunk Y: Y = (C.B^T o Lm).Xdt + exp(cum_i)*(C.prev^T) + D*x ----------------
__global__ void __launch_bounds__(256)
ydiag_k(const float* __restrict__ xBC, const float* __restrict__ dt_g,
        const float* __restrict__ cum, const float* __restrict__ prev,
        const float* __restrict__ Dp, float* __restrict__ Y) {
  const int blk = blockIdx.x;
  const int h = blk % H;
  const int c = (blk / H) % NCH;
  const int b = blk / (H * NCH);
  __shared__ __bf16 Cs[CS * Ns];       // [i][n]
  __shared__ __bf16 Bs[CS * Ns];       // [j][n]
  __shared__ __bf16 XdtT[P * CS];      // [p][j]
  __shared__ __bf16 Pv[P * Ns];        // prev [p][n]
  __shared__ float  cum_s[CS];
  __shared__ __bf16 Ssh[8 * 32 * 32];  // per-wave S tile staging

  const float* cumb = cum + ((size_t)(b * H + h) * NCH + c) * CS;
  {
    const int t = threadIdx.x;
    const int row = b * L + c * CS + t;
    cum_s[t] = cumb[t];
    const float dtv = dt_g[(size_t)row * H + h];
    const float* xr = xBC + (size_t)row * CONVD;
    for (int n = 0; n < Ns; ++n) {
      Bs[t * Ns + n] = (__bf16)xr[DI + n];
      Cs[t * Ns + n] = (__bf16)xr[DI + Ns + n];
    }
    for (int p = 0; p < P; ++p) XdtT[p * CS + t] = (__bf16)(xr[h * P + p] * dtv);
    const float* pb = prev + ((size_t)(b * NCH + c) * H + h) * (P * Ns);
    for (int e = t; e < P * Ns; e += 256) Pv[e] = (__bf16)pb[e];
  }
  __syncthreads();

  const int tid = threadIdx.x, lane = tid & 31, w = tid >> 5;
  const int lh = lane & 15, kb = (lane < 16) ? 0 : 8, kh = (lane < 16) ? 0 : 16;
  const int rbase = (lane < 16) ? 0 : 8;
  const int i0 = w * 32;

  v8f acc[2][4] = {};

  // Y_off = (C @ prev^T), then scale rows by exp(cum_i)
  for (int n0 = 0; n0 < Ns; n0 += 32) {
    v16bf a[2];
#pragma unroll
    for (int mi = 0; mi < 2; ++mi)
      a[mi] = load_a16(&Cs[(i0 + mi * 16 + lh) * Ns + n0], kb);
#pragma unroll
    for (int ni = 0; ni < 4; ++ni) {
      v16bf bf = *(const v16bf*)&Pv[(ni * 16 + lh) * Ns + n0 + kh];
#pragma unroll
      for (int mi = 0; mi < 2; ++mi)
        acc[mi][ni] = wmma_bf16(a[mi], bf, acc[mi][ni]);
    }
  }
#pragma unroll
  for (int mi = 0; mi < 2; ++mi) {
    float ei[8];
#pragma unroll
    for (int r = 0; r < 8; ++r) ei[r] = expf(cum_s[i0 + mi * 16 + r + rbase]);
#pragma unroll
    for (int ni = 0; ni < 4; ++ni)
#pragma unroll
      for (int r = 0; r < 8; ++r) acc[mi][ni][r] *= ei[r];
  }

  // lower-triangular j-blocks: T = C.B^T (K=64), mask+decay, then acc += S.Xdt (K=32)
  __bf16* Sw = &Ssh[w * 32 * 32];
  for (int jb = 0; jb <= w; ++jb) {
    const int j0 = jb * 32;
    v8f t2[2][2] = {};
    for (int n0 = 0; n0 < Ns; n0 += 32) {
      v16bf a[2];
#pragma unroll
      for (int mi = 0; mi < 2; ++mi)
        a[mi] = load_a16(&Cs[(i0 + mi * 16 + lh) * Ns + n0], kb);
#pragma unroll
      for (int tj = 0; tj < 2; ++tj) {
        v16bf bf = *(const v16bf*)&Bs[(j0 + tj * 16 + lh) * Ns + n0 + kh];
#pragma unroll
        for (int mi = 0; mi < 2; ++mi)
          t2[mi][tj] = wmma_bf16(a[mi], bf, t2[mi][tj]);
      }
    }
#pragma unroll
    for (int mi = 0; mi < 2; ++mi)
#pragma unroll
      for (int tj = 0; tj < 2; ++tj) {
        const int jidx = j0 + tj * 16 + lh;
#pragma unroll
        for (int r = 0; r < 8; ++r) {
          const int iidx = i0 + mi * 16 + r + rbase;
          const float v = (jidx <= iidx)
              ? t2[mi][tj][r] * expf(cum_s[iidx] - cum_s[jidx]) : 0.f;
          Sw[(mi * 16 + r + rbase) * 32 + (tj * 16 + lh)] = (__bf16)v;
        }
      }
    // per-wave LDS tile; DS ops are in-order per wave: safe to read back, no barrier
    v16bf sa[2];
#pragma unroll
    for (int mi = 0; mi < 2; ++mi)
      sa[mi] = load_a16(&Sw[(mi * 16 + lh) * 32], kb);
#pragma unroll
    for (int ni = 0; ni < 4; ++ni) {
      v16bf bf = *(const v16bf*)&XdtT[(ni * 16 + lh) * CS + j0 + kh];
#pragma unroll
      for (int mi = 0; mi < 2; ++mi)
        acc[mi][ni] = wmma_bf16(sa[mi], bf, acc[mi][ni]);
    }
  }

  const float Dh = Dp[h];
#pragma unroll
  for (int mi = 0; mi < 2; ++mi)
#pragma unroll
    for (int ni = 0; ni < 4; ++ni) {
      const int p = ni * 16 + lh;
#pragma unroll
      for (int r = 0; r < 8; ++r) {
        const int iidx = i0 + mi * 16 + r + rbase;
        const int row = b * L + c * CS + iidx;
        const float xv = xBC[(size_t)row * CONVD + h * P + p];
        Y[(size_t)row * DI + h * P + p] = acc[mi][ni][r] + Dh * xv;
      }
    }
}

// ---------------- gating (y * silu(z)) + RMSNorm(DI) -> bf16 ----------------
__global__ void __launch_bounds__(256)
gate_rms_k(const float* __restrict__ Y, const float* __restrict__ zx,
           const float* __restrict__ rms_w, __bf16* __restrict__ ybf) {
  const int row = blockIdx.x;
  const float* yr = Y + (size_t)row * DI;
  const float* zr = zx + (size_t)row * DIN;
  __shared__ float vsh[DI];
  __shared__ float red[256];
  float s = 0.f;
  for (int i = threadIdx.x; i < DI; i += 256) {
    const float z = zr[i];
    const float v = yr[i] * siluf(z);
    vsh[i] = v; s += v * v;
  }
  red[threadIdx.x] = s; __syncthreads();
  for (int off = 128; off > 0; off >>= 1) {
    if (threadIdx.x < off) red[threadIdx.x] += red[threadIdx.x + off];
    __syncthreads();
  }
  const float scale = rsqrtf(red[0] / (float)DI + EPS);
  for (int i = threadIdx.x; i < DI; i += 256)
    ybf[(size_t)row * DI + i] = (__bf16)(vsh[i] * scale * rms_w[i]);
}

// ---------------- launch ----------------
extern "C" void kernel_launch(void* const* d_in, const int* in_sizes, int n_in,
                              void* d_out, int out_size, void* d_ws, size_t ws_size,
                              hipStream_t stream) {
  (void)in_sizes; (void)n_in; (void)out_size; (void)ws_size;
  const float* u       = (const float*)d_in[0];
  const float* norm_w  = (const float*)d_in[1];
  const float* in_w    = (const float*)d_in[2];
  const float* conv_w  = (const float*)d_in[3];
  const float* conv_b  = (const float*)d_in[4];
  const float* dt_bias = (const float*)d_in[5];
  const float* A_log   = (const float*)d_in[6];
  const float* Dpw     = (const float*)d_in[7];
  const float* rms_w   = (const float*)d_in[8];
  const float* out_w   = (const float*)d_in[9];
  float* xout = (float*)d_out;

  char* ws = (char*)d_ws;
  auto take = [&](size_t bytes) { char* p = ws; ws += (bytes + 255) & ~(size_t)255; return p; };
  __bf16* xn_bf   = (__bf16*)take((size_t)ROWS * DM * 2);
  __bf16* wbf_in  = (__bf16*)take((size_t)DIN * DM * 2);
  __bf16* wbf_out = (__bf16*)take((size_t)DM * DI * 2);
  float*  zx      = (float*) take((size_t)ROWS * DIN * 4);
  float*  xBC     = (float*) take((size_t)ROWS * CONVD * 4);
  float*  dtg     = (float*) take((size_t)ROWS * H * 4);
  float*  cum     = (float*) take((size_t)Bb * H * NCH * CS * 4);
  float*  states  = (float*) take((size_t)Bb * NCH * H * P * Ns * 4);
  float*  prevs   = (float*) take((size_t)Bb * NCH * H * P * Ns * 4);
  float*  Ybuf    = (float*) take((size_t)ROWS * DI * 4);
  __bf16* ybf     = (__bf16*)take((size_t)ROWS * DI * 2);

  // residual accumulator starts as u
  hipMemcpyAsync(xout, u, (size_t)ROWS * DM * 4, hipMemcpyDeviceToDevice, stream);

  for (int layer = 0; layer < NL; ++layer) {
    {
      int n = DIN * DM;
      f32_to_bf16_k<<<(n + 255) / 256, 256, 0, stream>>>(in_w + (size_t)layer * DIN * DM, wbf_in, n);
      n = DM * DI;
      f32_to_bf16_k<<<(n + 255) / 256, 256, 0, stream>>>(out_w + (size_t)layer * DM * DI, wbf_out, n);
    }
    rmsnorm_to_bf16<<<ROWS, 256, 0, stream>>>(xout, norm_w + layer * DM, xn_bf);
    gemm_xwt<0><<<dim3(DIN / 128, ROWS / 128), 256, 0, stream>>>(xn_bf, wbf_in, zx, DIN, DM);
    {
      int n = ROWS * CONVD;
      conv_silu_k<<<(n + 255) / 256, 256, 0, stream>>>(zx, conv_w + layer * CONVD * DC,
                                                       conv_b + layer * CONVD, xBC);
    }
    dt_cum_k<<<Bb * H * NCH, 256, 0, stream>>>(zx, dt_bias + layer * H, A_log + layer * H, dtg, cum);
    states_k<<<Bb * NCH * H, 128, 0, stream>>>(xBC, dtg, cum, states);
    scan_k<<<Bb * H, 256, 0, stream>>>(states, cum, prevs);
    ydiag_k<<<Bb * NCH * H, 256, 0, stream>>>(xBC, dtg, cum, prevs, Dpw + layer * H, Ybuf);
    gate_rms_k<<<ROWS, 256, 0, stream>>>(Ybuf, zx, rms_w + layer * DI, ybf);
    gemm_xwt<1><<<dim3(DM / 128, ROWS / 128), 256, 0, stream>>>(ybf, wbf_out, xout, DM, DI);
  }
}
